// xLSTMModel_60636348285650
// MI455X (gfx1250) — matrix operand order repaired and checked by reading, not compiled, not run on hardware
//
#include <hip/hip_runtime.h>
#include <hip/hip_bf16.h>

typedef __attribute__((ext_vector_type(16))) _Float16 v16h;
typedef __attribute__((ext_vector_type(8)))  _Float16 v8h;
typedef __attribute__((ext_vector_type(8)))  float    v8f;

#define Bn     128
#define Sn     100
#define INn    64
#define HIDn   1024
#define NHn    8
#define INNERn 2048
#define FFn    1344
#define OUTn   64
#define BSn    (Bn*Sn)

// ---------------------------------------------------------------- helpers
__device__ __forceinline__ float logsigf(float x) {
    return (x >= 0.f) ? -log1pf(expf(-x)) : x - log1pf(expf(x));
}
__device__ __forceinline__ float siluf(float x) { return x / (1.f + expf(-x)); }
__device__ __forceinline__ float sigmf(float x) { return 1.f / (1.f + expf(-x)); }

// Build a 16x32 f16 fragment from two contiguous 8-half runs in LDS (f16 source).
__device__ __forceinline__ v16h frag16(const _Float16* p0, const _Float16* p1) {
    v8h a = *(const v8h*)p0;
    v8h b = *(const v8h*)p1;
    v16h r;
#pragma unroll
    for (int i = 0; i < 8; i++) { r[i] = a[i]; r[i + 8] = b[i]; }
    return r;
}

// Build a 16x32 f16 fragment from two contiguous 8-float runs in LDS (f32 source,
// converted on the fly; VALU converts co-execute with the matrix pipe).
__device__ __forceinline__ v16h fragf(const float* p0, const float* p1) {
    v16h r;
#pragma unroll
    for (int i = 0; i < 8; i++) { r[i] = (_Float16)p0[i]; r[i + 8] = (_Float16)p1[i]; }
    return r;
}

// CDNA5 async global->LDS copy of 16 bytes (tracked by ASYNCcnt).
// low 32 bits of the flat shared-memory pointer are the
// wave-relative LDS byte offset (flat LDS aperture keeps the offset in addr[31:0]).
__device__ __forceinline__ void async_cp16(void* lds_dst, const void* gsrc) {
    unsigned l = (unsigned)(unsigned long long)(uintptr_t)lds_dst;
    unsigned long long g = (unsigned long long)(uintptr_t)gsrc;
    asm volatile("global_load_async_to_lds_b128 %0, %1, off"
                 :: "v"(l), "v"(g) : "memory");
}
__device__ __forceinline__ void async_wait0() {
    asm volatile("s_wait_asynccnt 0" ::: "memory");
}

// ---------------------------------------------------------------- generic WMMA GEMM
// OUT[m, col] (+)= sum_k A[m*lda + k] * W[col*K + k]  (+ bias[col])
// block = 256 threads (8 waves); block tile 64(M) x 64(N); wave -> 16x32 strip
// (2 wmma / k-step, shared A fragment). Double-buffered LDS staged with
// global_load_async_to_lds_b128; f32 tiles converted to f16 at fragment build.
// Requirements (all call sites): M%64==0, N%64==0, K%32==0.
__global__ void k_gemm(const float* __restrict__ A, long long lda,
                       const float* __restrict__ W,
                       const float* __restrict__ bias,
                       float* __restrict__ OUT, long long ldo,
                       int M, int N, int K, int accum)
{
    __shared__ __align__(16) float sA[2][64 * 36];  // 64 rows x 32 f32, stride 36
    __shared__ __align__(16) float sB[2][64 * 36];

    const int tid  = threadIdx.x;
    const int wave = tid >> 5;
    const int lane = tid & 31;
    const long long bm0 = (long long)blockIdx.x * 64;
    const long long bn0 = (long long)blockIdx.y * 64;
    const int mt  = wave & 3;   // 0..3  (M subtile)
    const int nt  = wave >> 2;  // 0..1  (N strip of 32)
    const int g   = lane >> 4;  // half-wave
    const int l16 = lane & 15;

    v8f acc0 = {}, acc1 = {};

    // prologue: stage k0 = 0 into buffer 0 (512 A chunks + 512 W chunks, 2+2/thread)
    for (int c = tid; c < 512; c += 256) {
        int m = c >> 3, p = c & 7;                      // row, 16B chunk in row
        async_cp16(&sA[0][m * 36 + p * 4], A + (bm0 + m) * lda + p * 4);
        async_cp16(&sB[0][m * 36 + p * 4], W + (bn0 + m) * (long long)K + p * 4);
    }
    async_wait0();
    __syncthreads();

    int buf = 0;
    for (int k0 = 0; k0 < K; k0 += 32) {
        if (k0 + 32 < K) {  // stage next k-tile into the other buffer (overlaps WMMA)
            for (int c = tid; c < 512; c += 256) {
                int m = c >> 3, p = c & 7;
                async_cp16(&sA[buf ^ 1][m * 36 + p * 4],
                           A + (bm0 + m) * lda + (k0 + 32) + p * 4);
                async_cp16(&sB[buf ^ 1][m * 36 + p * 4],
                           W + (bn0 + m) * (long long)K + (k0 + 32) + p * 4);
            }
        }
        const float* sa = &sA[buf][(mt * 16 + l16) * 36];
        v16h a = fragf(sa + g * 16, sa + g * 16 + 8);
        const float* sb0 = &sB[buf][(nt * 32 + l16) * 36];
        v16h b0 = fragf(sb0 + g * 16, sb0 + g * 16 + 8);
        acc0 = __builtin_amdgcn_wmma_f32_16x16x32_f16(false, a, false, b0,
                                                      (short)0, acc0, false, false);
        const float* sb1 = &sB[buf][(nt * 32 + 16 + l16) * 36];
        v16h b1 = fragf(sb1 + g * 16, sb1 + g * 16 + 8);
        acc1 = __builtin_amdgcn_wmma_f32_16x16x32_f16(false, a, false, b1,
                                                      (short)0, acc1, false, false);
        async_wait0();
        __syncthreads();
        buf ^= 1;
    }

    const long long mbase = bm0 + mt * 16 + g * 8;
    const long long col0  = bn0 + nt * 32 + l16;
    const long long col1  = col0 + 16;
    const float bv0 = bias ? bias[col0] : 0.f;
    const float bv1 = bias ? bias[col1] : 0.f;
#pragma unroll
    for (int r = 0; r < 8; r++) {
        long long m = mbase + r;
        float v0 = acc0[r] + bv0;
        float v1 = acc1[r] + bv1;
        if (accum) { v0 += OUT[m * ldo + col0]; v1 += OUT[m * ldo + col1]; }
        OUT[m * ldo + col0] = v0;
        OUT[m * ldo + col1] = v1;
    }
}

// ---------------------------------------------------------------- layernorm (per row)
__global__ void k_ln(const float* __restrict__ in, const float* __restrict__ w,
                     float* __restrict__ out, int D)
{
    __shared__ float red[256];
    const long long row = blockIdx.x;
    const float* xr = in + row * (long long)D;
    float s = 0.f, ss = 0.f;
    for (int j = threadIdx.x; j < D; j += 256) { float x = xr[j]; s += x; ss += x * x; }
    red[threadIdx.x] = s; __syncthreads();
    for (int o = 128; o; o >>= 1) { if (threadIdx.x < o) red[threadIdx.x] += red[threadIdx.x + o]; __syncthreads(); }
    float mu = red[0] / D; __syncthreads();
    red[threadIdx.x] = ss; __syncthreads();
    for (int o = 128; o; o >>= 1) { if (threadIdx.x < o) red[threadIdx.x] += red[threadIdx.x + o]; __syncthreads(); }
    float var  = red[0] / D - mu * mu;
    float rstd = rsqrtf(var + 1e-5f);
    for (int j = threadIdx.x; j < D; j += 256)
        out[row * (long long)D + j] = (xr[j] - mu) * rstd * w[j];
}

// ---------------------------------------------------------------- causal depthwise conv (K=4) + SiLU
__global__ void k_conv_silu(const float* __restrict__ x, long long ldx,
                            const float* __restrict__ w, const float* __restrict__ bias,
                            float* __restrict__ out, long long ldo,
                            int B, int S, int D)
{
    long long idx = (long long)blockIdx.x * 256 + threadIdx.x;
    long long total = (long long)B * S * D;
    if (idx >= total) return;
    int d = (int)(idx % D);
    long long bs = idx / D;
    int s = (int)(bs % S);
    long long b = bs / S;
    float acc = bias[d];
#pragma unroll
    for (int kk = 0; kk < 4; kk++) {
        int sp = s - 3 + kk;
        if (sp >= 0) acc += x[((long long)b * S + sp) * ldx + d] * w[d * 4 + kk];
    }
    out[bs * ldo + d] = siluf(acc);
}

// ---------------------------------------------------------------- 4x4 block-diagonal projection (q/k/v)
__global__ void k_headwise4(const float* __restrict__ x, long long ldx,
                            const float* __restrict__ w, float* __restrict__ out,
                            long long n)
{
    long long idx = (long long)blockIdx.x * 256 + threadIdx.x;
    if (idx >= n) return;
    int c = (int)(idx % 2048);
    long long bs = idx / 2048;
    int nb = c >> 2, o = c & 3;
    const float* xr = x + bs * ldx + nb * 4;
    const float* wr = w + nb * 16 + o * 4;
    out[bs * 2048 + c] = xr[0]*wr[0] + xr[1]*wr[1] + xr[2]*wr[2] + xr[3]*wr[3];
}

// ---------------------------------------------------------------- ig/fg projections (N=8, too thin for WMMA)
__global__ void k_igfg(const float* __restrict__ q, const float* __restrict__ k,
                       const float* __restrict__ v,
                       const float* __restrict__ wig, const float* __restrict__ big,
                       const float* __restrict__ wfg, const float* __restrict__ bfg,
                       float* __restrict__ ig, float* __restrict__ fg)
{
    long long bs = blockIdx.x;
    int w = threadIdx.x >> 5, lane = threadIdx.x & 31;
    const float* qr = q + bs * 2048;
    const float* kr = k + bs * 2048;
    const float* vr = v + bs * 2048;
    const float* wi = wig + (long long)w * 6144;
    const float* wf = wfg + (long long)w * 6144;
    float ai = 0.f, af = 0.f;
    for (int j = lane; j < 2048; j += 32) { float x = qr[j]; ai += x * wi[j];        af += x * wf[j]; }
    for (int j = lane; j < 2048; j += 32) { float x = kr[j]; ai += x * wi[2048 + j]; af += x * wf[2048 + j]; }
    for (int j = lane; j < 2048; j += 32) { float x = vr[j]; ai += x * wi[4096 + j]; af += x * wf[4096 + j]; }
    for (int o = 16; o; o >>= 1) { ai += __shfl_xor(ai, o, 32); af += __shfl_xor(af, o, 32); }
    if (lane == 0) { ig[bs * 8 + w] = ai + big[w]; fg[bs * 8 + w] = af + bfg[w]; }
}

// ---------------------------------------------------------------- mLSTM parallel attention (per b,head)
__global__ void k_mlstm(const float* __restrict__ q, const float* __restrict__ k,
                        const float* __restrict__ v,
                        const float* __restrict__ ig, const float* __restrict__ fg,
                        float* __restrict__ hc, int S)
{
    const int b = blockIdx.x >> 3;
    const int h = blockIdx.x & 7;
    const int tid = threadIdx.x;
    __shared__ float lfcs[128], igv[128], qv[256], sco[128], slg[128], red[256];
    const long long bofs = (long long)b * S;
    const long long hofs = (long long)h * 256;

    for (int s = tid; s < S; s += 256) {
        lfcs[s] = logsigf(fg[(bofs + s) * 8 + h]);
        igv[s]  = ig[(bofs + s) * 8 + h];
    }
    __syncthreads();
    if (tid == 0) { float c = 0.f; for (int s = 0; s < S; s++) { c += lfcs[s]; lfcs[s] = c; } }
    __syncthreads();

    const float scale = 0.0625f; // 256^-0.5
    for (int sq = 0; sq < S; sq++) {
        qv[tid] = q[(bofs + sq) * 2048 + hofs + tid];
        __syncthreads();
        float lm = -3.4e38f;
        for (int t = tid; t <= sq; t += 256) {
            const float* kr = k + (bofs + t) * 2048 + hofs;
            float dot = 0.f;
            for (int d = 0; d < 256; d++) dot += qv[d] * kr[d];
            sco[t] = dot;
            float lg = lfcs[sq] - lfcs[t] + igv[t];
            slg[t] = lg;
            lm = fmaxf(lm, lg);
        }
        red[tid] = lm; __syncthreads();
        for (int o = 128; o; o >>= 1) { if (tid < o) red[tid] = fmaxf(red[tid], red[tid + o]); __syncthreads(); }
        float maxD = red[0]; __syncthreads();
        float lsum = 0.f;
        for (int t = tid; t <= sq; t += 256) {
            float c = sco[t] * scale * expf(slg[t] - maxD);
            sco[t] = c; lsum += c;
        }
        red[tid] = lsum; __syncthreads();
        for (int o = 128; o; o >>= 1) { if (tid < o) red[tid] += red[tid + o]; __syncthreads(); }
        float norm = fmaxf(fabsf(red[0]), expf(-maxD)) + 1e-6f;
        __syncthreads();
        {
            int d = tid; // DH == blockDim == 256
            float acc = 0.f;
            for (int t = 0; t <= sq; t++) acc += sco[t] * v[(bofs + t) * 2048 + hofs + d];
            hc[(bofs + sq) * 2048 + hofs + d] = acc / norm;
        }
        __syncthreads();
    }
}

// ---------------------------------------------------------------- mLSTM out-norm + skip + SiLU(z) gate  (hc -> hs, in place)
__global__ void k_mhln_gate(float* __restrict__ hc, const float* __restrict__ onw,
                            const float* __restrict__ skipw, const float* __restrict__ xc,
                            const float* __restrict__ xi /* z at col 2048.. */)
{
    long long bs = blockIdx.x;
    int w = threadIdx.x >> 5, lane = threadIdx.x & 31;
    size_t base = (size_t)bs * 2048 + w * 256;
    float s = 0.f, ss = 0.f;
    for (int d = lane; d < 256; d += 32) { float x = hc[base + d]; s += x; ss += x * x; }
    for (int o = 16; o; o >>= 1) { s += __shfl_xor(s, o, 32); ss += __shfl_xor(ss, o, 32); }
    float mu = s / 256.f, var = ss / 256.f - mu * mu;
    float rstd = rsqrtf(var + 1e-5f);
    for (int d = lane; d < 256; d += 32) {
        int ch = w * 256 + d;
        float hn = (hc[base + d] - mu) * rstd * onw[ch];
        float zz = xi[(size_t)bs * 4096 + 2048 + ch];
        hc[base + d] = (hn + skipw[ch] * xc[(size_t)bs * 2048 + ch]) * siluf(zz);
    }
}

// ---------------------------------------------------------------- sLSTM scan, recurrent matmul via WMMA
// grid: (head 0..7, bchunk 0..3); block 256 threads (8 waves); batch chunk = 32
__global__ void k_slstm(const float* __restrict__ Wx,   // (B,S,4096)
                        const float* __restrict__ R,    // (8,128,512)
                        const float* __restrict__ sb,   // (4096)
                        float* __restrict__ ys,         // (B,S,1024)
                        float* __restrict__ stC, float* __restrict__ stN,
                        float* __restrict__ stM,
                        _Float16* __restrict__ y16,     // (8,128,128) [h][b][d]
                        float* __restrict__ ryws, int S)
{
    __shared__ __align__(16) _Float16 sA[32 * 136];
    __shared__ __align__(16) _Float16 sB[64 * 136];

    const int h  = blockIdx.x;
    const int b0 = blockIdx.y * 32;
    const int tid = threadIdx.x, wave = tid >> 5, lane = tid & 31;
    const int g = lane >> 4, l16 = lane & 15;
    float* ry = ryws + (size_t)(blockIdx.y * 8 + h) * (32 * 512);
    _Float16* yh = y16 + (size_t)h * 128 * 128 + (size_t)b0 * 128;

    // init per-(b,h) state
    for (int e = tid; e < 32 * 128; e += 256) {
        int bl = e >> 7, d = e & 127;
        size_t si = (size_t)(b0 + bl) * 1024 + h * 128 + d;
        stC[si] = 0.f; stN[si] = 0.f; stM[si] = 0.f;
        yh[bl * 128 + d] = (_Float16)0.f;
    }
    __syncthreads();

    for (int s = 0; s < S; s++) {
        // stage y_f16 (32 x 128) for this step
        for (int e = tid; e < 32 * 128; e += 256) {
            int m = e >> 7, kk = e & 127;
            sA[m * 136 + kk] = yh[m * 128 + kk];
        }
        __syncthreads();
        // GEMM  Ry = Y(32x128) * Rh(128x512), in 8 N-chunks of 64
        for (int ch = 0; ch < 8; ch++) {
            for (int e = tid; e < 64 * 128; e += 256) {
                int nn = e & 63, kk = e >> 6;
                sB[nn * 136 + kk] =
                    (_Float16)R[(size_t)h * 65536 + (size_t)kk * 512 + ch * 64 + nn];
            }
            __syncthreads();
            const int mt = wave & 1, nt = wave >> 1;
            v8f acc = {};
#pragma unroll
            for (int k0 = 0; k0 < 128; k0 += 32) {
                const _Float16* pa = &sA[(mt * 16 + l16) * 136 + k0 + g * 16];
                v16h a = frag16(pa, pa + 8);
                const _Float16* pb = &sB[(nt * 16 + l16) * 136 + k0 + g * 16];
                v16h b = frag16(pb, pb + 8);
                acc = __builtin_amdgcn_wmma_f32_16x16x32_f16(false, a, false, b,
                                                             (short)0, acc, false, false);
            }
            const int col   = ch * 64 + nt * 16 + l16;
            const int mbase = mt * 16 + g * 8;
#pragma unroll
            for (int r2 = 0; r2 < 8; r2++) ry[(size_t)(mbase + r2) * 512 + col] = acc[r2];
            __syncthreads();
        }
        // gate phase
        for (int e = tid; e < 32 * 128; e += 256) {
            int bl = e >> 7, d = e & 127;
            int b = b0 + bl;
            int hch = h * 128 + d;
            size_t wbase = ((size_t)b * S + s) * 4096;
            float ir  = Wx[wbase + hch]        + ry[(size_t)bl * 512 + d]       + sb[hch];
            float fr  = Wx[wbase + 1024 + hch] + ry[(size_t)bl * 512 + 128 + d] + sb[1024 + hch];
            float zr  = Wx[wbase + 2048 + hch] + ry[(size_t)bl * 512 + 256 + d] + sb[2048 + hch];
            float orr = Wx[wbase + 3072 + hch] + ry[(size_t)bl * 512 + 384 + d] + sb[3072 + hch];
            size_t si = (size_t)b * 1024 + hch;
            float lfpm = stM[si] + logsigf(fr);
            float mnew = (s == 0) ? ir : fmaxf(ir, lfpm);
            float ig_  = expf(ir - mnew);
            float fg_  = expf(lfpm - mnew);
            float cnew = fg_ * stC[si] + ig_ * tanhf(zr);
            float nnew = fg_ * stN[si] + ig_;
            float ynew = sigmf(orr) * cnew / nnew;
            stC[si] = cnew; stN[si] = nnew; stM[si] = mnew;
            yh[bl * 128 + d] = (_Float16)ynew;
            ys[((size_t)b * S + s) * 1024 + hch] = ynew;
        }
        __syncthreads();
    }
}

// ---------------------------------------------------------------- sLSTM group norm + residual into h
__global__ void k_sgn_res(const float* __restrict__ ys, const float* __restrict__ w,
                          float* __restrict__ h)
{
    long long bs = blockIdx.x;
    int wv = threadIdx.x >> 5, lane = threadIdx.x & 31;
    size_t base = (size_t)bs * 1024 + wv * 128;
    float s = 0.f, ss = 0.f;
    for (int d = lane; d < 128; d += 32) { float x = ys[base + d]; s += x; ss += x * x; }
    for (int o = 16; o; o >>= 1) { s += __shfl_xor(s, o, 32); ss += __shfl_xor(ss, o, 32); }
    float mu = s / 128.f, var = ss / 128.f - mu * mu;
    float rstd = rsqrtf(var + 1e-5f);
    for (int d = lane; d < 128; d += 32) {
        int ch = wv * 128 + d;
        h[base + d] += (ys[base + d] - mu) * rstd * w[ch];
    }
}

// ---------------------------------------------------------------- FFN gate: gelu(up[:FF]) * up[FF:]
__global__ void k_gelu_gate(const float* __restrict__ up, float* __restrict__ out, long long n)
{
    long long idx = (long long)blockIdx.x * 256 + threadIdx.x;
    if (idx >= n) return;
    int j = (int)(idx % FFn);
    long long bs = idx / FFn;
    float gx = up[bs * 2688 + j];
    float t  = up[bs * 2688 + FFn + j];
    out[idx] = 0.5f * gx * (1.f + erff(gx * 0.70710678118654752f)) * t;
}

// ---------------------------------------------------------------- launch
extern "C" void kernel_launch(void* const* d_in, const int* in_sizes, int n_in,
                              void* d_out, int out_size, void* d_ws, size_t ws_size,
                              hipStream_t stream)
{
    const float* x        = (const float*)d_in[0];
    const float* emb_w    = (const float*)d_in[1];
    const float* emb_b    = (const float*)d_in[2];
    const float* ln0_w    = (const float*)d_in[3];
    const float* m_up_w   = (const float*)d_in[4];
    const float* m_conv_w = (const float*)d_in[5];
    const float* m_conv_b = (const float*)d_in[6];
    const float* m_q_w    = (const float*)d_in[7];
    const float* m_k_w    = (const float*)d_in[8];
    const float* m_v_w    = (const float*)d_in[9];
    const float* m_ig_w   = (const float*)d_in[10];
    const float* m_ig_b   = (const float*)d_in[11];
    const float* m_fg_w   = (const float*)d_in[12];
    const float* m_fg_b   = (const float*)d_in[13];
    const float* m_on_w   = (const float*)d_in[14];
    const float* m_skip   = (const float*)d_in[15];
    const float* m_down_w = (const float*)d_in[16];
    const float* ln1_w    = (const float*)d_in[17];
    const float* s_conv_w = (const float*)d_in[18];
    const float* s_conv_b = (const float*)d_in[19];
    const float* s_i_w    = (const float*)d_in[20];
    const float* s_f_w    = (const float*)d_in[21];
    const float* s_z_w    = (const float*)d_in[22];
    const float* s_o_w    = (const float*)d_in[23];
    const float* s_R      = (const float*)d_in[24];
    const float* s_b      = (const float*)d_in[25];
    const float* s_gn_w   = (const float*)d_in[26];
    const float* ffn_ln_w = (const float*)d_in[27];
    const float* ff_up_w  = (const float*)d_in[28];
    const float* ff_down_w= (const float*)d_in[29];
    const float* post_ln_w= (const float*)d_in[30];
    const float* fc_w     = (const float*)d_in[31];
    const float* fc_b     = (const float*)d_in[32];

    // ---- workspace arena (liveness-based reuse) ----
    char* base = (char*)d_ws;
    size_t off = 0;
    auto alloc = [&](size_t nbytes) -> void* {
        void* p = base + off;
        off += (nbytes + 255) & ~(size_t)255;
        return p;
    };
    float* h   = (float*)alloc((size_t)BSn * HIDn * 4);
    float* r   = (float*)alloc((size_t)BSn * HIDn * 4);
    float* xi  = (float*)alloc((size_t)BSn * 4096 * 4);   // xi -> Wx -> up
    float* xc  = (float*)alloc((size_t)BSn * 2048 * 4);   // x_c -> gated
    float* qb  = (float*)alloc((size_t)BSn * 2048 * 4);   // q -> rc
    float* kb  = (float*)alloc((size_t)BSn * 2048 * 4);   // k -> ys
    float* vb  = (float*)alloc((size_t)BSn * 2048 * 4);
    float* hc  = (float*)alloc((size_t)BSn * 2048 * 4);   // hc -> hs (in place)
    float* igb = (float*)alloc((size_t)BSn * 8 * 4);
    float* fgb = (float*)alloc((size_t)BSn * 8 * 4);
    float* stC = (float*)alloc((size_t)Bn * HIDn * 4);
    float* stN = (float*)alloc((size_t)Bn * HIDn * 4);
    float* stM = (float*)alloc((size_t)Bn * HIDn * 4);
    _Float16* y16 = (_Float16*)alloc((size_t)8 * 128 * 128 * 2);
    float* ryws = (float*)alloc((size_t)32 * 32 * 512 * 4);

    dim3 blk(256);

    // 1) embedding: h = x @ emb_w.T + emb_b
    k_gemm<<<dim3(BSn/64, HIDn/64), blk, 0, stream>>>(x, INn, emb_w, emb_b, h, HIDn,
                                                      BSn, HIDn, INn, 0);
    // 2) r = ln0(h)
    k_ln<<<BSn, blk, 0, stream>>>(h, ln0_w, r, HIDn);
    // 3) xi = r @ m_up_w.T   (B,S,4096)
    k_gemm<<<dim3(BSn/64, 4096/64), blk, 0, stream>>>(r, HIDn, m_up_w, nullptr, xi, 4096,
                                                      BSn, 4096, HIDn, 0);
    // 4) x_c = silu(causal_conv(x_m))   (x_m = xi[:, :2048])
    long long tot2048 = (long long)BSn * 2048;
    k_conv_silu<<<(unsigned)((tot2048 + 255) / 256), blk, 0, stream>>>(
        xi, 4096, m_conv_w, m_conv_b, xc, 2048, Bn, Sn, 2048);
    // 5) q, k from x_c ; v from x_m
    k_headwise4<<<(unsigned)((tot2048 + 255) / 256), blk, 0, stream>>>(xc, 2048, m_q_w, qb, tot2048);
    k_headwise4<<<(unsigned)((tot2048 + 255) / 256), blk, 0, stream>>>(xc, 2048, m_k_w, kb, tot2048);
    k_headwise4<<<(unsigned)((tot2048 + 255) / 256), blk, 0, stream>>>(xi, 4096, m_v_w, vb, tot2048);
    // 6) ig, fg
    k_igfg<<<BSn, blk, 0, stream>>>(qb, kb, vb, m_ig_w, m_ig_b, m_fg_w, m_fg_b, igb, fgb);
    // 7) mLSTM parallel attention -> hc
    k_mlstm<<<Bn * NHn, blk, 0, stream>>>(qb, kb, vb, igb, fgb, hc, Sn);
    // 8) hs = (mh_ln(hc)*on_w + skip*x_c) * silu(z)   (in place in hc)
    k_mhln_gate<<<BSn, blk, 0, stream>>>(hc, m_on_w, m_skip, xc, xi);
    // 9) h += hs @ m_down_w.T
    k_gemm<<<dim3(BSn/64, HIDn/64), blk, 0, stream>>>(hc, 2048, m_down_w, nullptr, h, HIDn,
                                                      BSn, HIDn, 2048, 1);
    // 10) r = ln1(h)
    k_ln<<<BSn, blk, 0, stream>>>(h, ln1_w, r, HIDn);
    // 11) rc = silu(causal_conv(r))   (rc reuses q buffer)
    long long tot1024 = (long long)BSn * 1024;
    k_conv_silu<<<(unsigned)((tot1024 + 255) / 256), blk, 0, stream>>>(
        r, 1024, s_conv_w, s_conv_b, qb, 1024, Bn, Sn, 1024);
    // 12) Wx = concat(headwise(rc,i), headwise(rc,f), headwise(r,z), headwise(r,o))  -> xi (B,S,4096)
    {
        const float* gsrc[4] = { qb, qb, r, r };
        const float* gw[4]   = { s_i_w, s_f_w, s_z_w, s_o_w };
        for (int gidx = 0; gidx < 4; gidx++)
            for (int hh = 0; hh < 8; hh++)
                k_gemm<<<dim3(BSn/64, 128/64), blk, 0, stream>>>(
                    gsrc[gidx] + hh * 128, 1024,
                    gw[gidx] + (size_t)hh * 128 * 128, nullptr,
                    xi + gidx * 1024 + hh * 128, 4096,
                    BSn, 128, 128, 0);
    }
    // 13) sLSTM scan -> ys (reuses k buffer); recurrent matmul via WMMA
    k_slstm<<<dim3(8, 4), blk, 0, stream>>>(xi, s_R, s_b, kb, stC, stN, stM, y16, ryws, Sn);
    // 14) h += mh_ln(ys) * s_gn_w
    k_sgn_res<<<BSn, blk, 0, stream>>>(kb, s_gn_w, h);
    // 15) r = ln(h)
    k_ln<<<BSn, blk, 0, stream>>>(h, ffn_ln_w, r, HIDn);
    // 16) up = r @ ff_up_w.T   (reuses xi, ld 2688)
    k_gemm<<<dim3(BSn/64, 2688/64), blk, 0, stream>>>(r, 1024, ff_up_w, nullptr, xi, 2688,
                                                      BSn, 2688, 1024, 0);
    // 17) gated = gelu(up[:FF]) * up[FF:]   (reuses xc, ld 1344)
    long long totff = (long long)BSn * FFn;
    k_gelu_gate<<<(unsigned)((totff + 255) / 256), blk, 0, stream>>>(xi, xc, totff);
    // 18) h += gated @ ff_down_w.T
    k_gemm<<<dim3(BSn/64, HIDn/64), blk, 0, stream>>>(xc, FFn, ff_down_w, nullptr, h, HIDn,
                                                      BSn, HIDn, FFn, 1);
    // 19) r = post_ln(h)
    k_ln<<<BSn, blk, 0, stream>>>(h, post_ln_w, r, HIDn);
    // 20) out = r[:, -1, :] @ fc_w.T + fc_b   (stride trick selects s = S-1)
    k_gemm<<<dim3(Bn/64, OUTn/64), blk, 0, stream>>>(r + (Sn - 1) * HIDn, (long long)Sn * HIDn,
                                                     fc_w, fc_b, (float*)d_out, OUTn,
                                                     Bn, OUTn, HIDn, 0);
}
